// Attention_81612968559134
// MI455X (gfx1250) — compile-verified, matrix operand order checked by reading, not run on hardware
//
#include <hip/hip_runtime.h>
#include <hip/hip_bf16.h>

// ---------------------------------------------------------------------------
// Problem constants (from reference): B=2, S=2048, D=768, H=12, DH=64
// ---------------------------------------------------------------------------
#define BB 2
#define SS 2048
#define DD 768
#define HH 12
#define DHD 64
#define MTOT (BB * SS)          // 4096 token rows
#define NQK (HH * DHD)          // 768
#define SCALE 0.125f            // DH^-0.5 = 1/8

typedef __attribute__((ext_vector_type(16))) _Float16 v16h;
typedef __attribute__((ext_vector_type(8)))  float    v8f;

// ---------------------------------------------------------------------------
// WMMA wrapper: D = A(16x32 f16) * B(32x16 f16) + C(16x16 f32)
// ---------------------------------------------------------------------------
__device__ __forceinline__ v8f wmma_f16(v16h a, v16h b, v8f c) {
  return __builtin_amdgcn_wmma_f32_16x16x32_f16(
      /*neg_a=*/false, a, /*neg_b=*/false, b,
      /*c_mod=*/(short)0, c, /*reuse_a=*/false, /*reuse_b=*/false);
}

// ---------------------------------------------------------------------------
// Load a 16x32 f16 fragment (A layout; same indexing serves B when the source
// is stored contraction-dim-contiguous, i.e. [n][k]).
// CDNA5 wave32 A layout: lanes 0-15 -> M=lane,    K in {0..7, 16..23}
//                        lanes 16-31 -> M=lane-16, K in {8..15, 24..31}
// K-pairs are contiguous -> compiler merges into ds_load_b128 pairs.
// ---------------------------------------------------------------------------
__device__ __forceinline__ v16h load_frag(const _Float16* p, int ld) {
  int lane = threadIdx.x & 31;
  int l15  = lane & 15;
  int kofs = (lane & 16) ? 8 : 0;
  const _Float16* q = p + l15 * ld + kofs;
  v16h a;
#pragma unroll
  for (int i = 0; i < 4; ++i) { a[2*i]   = q[2*i];    a[2*i+1] = q[2*i+1]; }
#pragma unroll
  for (int i = 0; i < 4; ++i) { a[8+2*i] = q[16+2*i]; a[9+2*i] = q[16+2*i+1]; }
  return a;
}

// ---------------------------------------------------------------------------
// f32 -> f16 conversion of X
// ---------------------------------------------------------------------------
__global__ __launch_bounds__(256) void cvt_x(const float* __restrict__ x,
                                             _Float16* __restrict__ o, int n) {
  int i = (blockIdx.x * 256 + threadIdx.x) * 4;
  if (i < n) {
    float4 v = *(const float4*)(x + i);
    _Float16 h[4] = {(_Float16)v.x, (_Float16)v.y, (_Float16)v.z, (_Float16)v.w};
    *(uint2*)(o + i) = *(const uint2*)h;
  }
}

// ---------------------------------------------------------------------------
// Convert + transpose weights: Wt[w][n][k] = W_w[k][n]  (w = q,k,v,o)
// ---------------------------------------------------------------------------
__global__ __launch_bounds__(256) void cvt_w(const float* __restrict__ wq,
                                             const float* __restrict__ wk,
                                             const float* __restrict__ wv,
                                             const float* __restrict__ wo,
                                             _Float16* __restrict__ Wt) {
  int i = blockIdx.x * 256 + threadIdx.x;
  if (i < 4 * DD * DD) {
    int w   = i / (DD * DD);
    int rem = i - w * (DD * DD);
    int n   = rem / DD;
    int k   = rem - n * DD;
    const float* W = (w == 0) ? wq : (w == 1) ? wk : (w == 2) ? wv : wo;
    Wt[i] = (_Float16)W[k * DD + n];
  }
}

// ---------------------------------------------------------------------------
// RoPE table: tab[s*32+d] = (cos, sin)(s * 10000^(-d/32)).
// 10000^(-d/32) = e^(-d * ln(10000)/32) -> single v_exp_f32 via __expf.
// ---------------------------------------------------------------------------
__global__ __launch_bounds__(256) void rope_tab_k(float2* __restrict__ tab) {
  int i = blockIdx.x * 256 + threadIdx.x;
  if (i < SS * 32) {
    int s = i >> 5, d = i & 31;
    float invf = __expf(-(float)d * (9.210340371976184f / 32.0f));
    float sn, cs;
    __sincosf((float)s * invf, &sn, &cs);
    tab[i] = make_float2(cs, sn);
  }
}

// ---------------------------------------------------------------------------
// QKV projection GEMM + fused RoPE (table-driven), software-pipelined.
// grid = (M/128, N/128, 3)  z: 0=Q(rope) 1=K(rope) 2=V
// block = 256 (8 waves) arranged 4x2: wave computes 32 rows x 64 cols
//   -> per K-step per wave: 2 A-frags + 4 B-frags feed 8 WMMAs.
// Outputs f16 head-major [B][H][S][DH].
// ---------------------------------------------------------------------------
__global__ __launch_bounds__(256) void gemm_qkv_rope(
    const _Float16* __restrict__ X,    // [4096][768]
    const _Float16* __restrict__ WtA,  // [4][768 n][768 k]
    const float2* __restrict__ rtab,   // [2048][32] (cos,sin)
    _Float16* __restrict__ Qo, _Float16* __restrict__ Ko,
    _Float16* __restrict__ Vo) {
  const int LDP = 40;
  __shared__ _Float16 Xs[128 * LDP];
  __shared__ _Float16 Ws[128 * LDP];

  int m0   = blockIdx.x * 128;
  int n0   = blockIdx.y * 128;
  int z    = blockIdx.z;
  const _Float16* W = WtA + (size_t)z * DD * DD;
  int tid  = threadIdx.x;
  int wave = tid >> 5, lane = tid & 31;
  int l15  = lane & 15;
  int rbase = (lane & 16) ? 8 : 0;
  int mq = (wave & 3) * 32;   // wave row offset within block
  int nq = (wave >> 2) * 64;  // wave col offset within block

  // staging-register addresses (fixed per thread)
  int er = (tid * 8) >> 5, ec = (tid * 8) & 31;          // pass 0
  int er2 = ((tid + 256) * 8) >> 5, ec2 = ((tid + 256) * 8) & 31;  // pass 1

  uint4 xr0, xr1, wr0, wr1;
  auto load_regs = [&](int kb) {
    xr0 = *(const uint4*)(X + (size_t)(m0 + er)  * DD + kb + ec);
    xr1 = *(const uint4*)(X + (size_t)(m0 + er2) * DD + kb + ec2);
    wr0 = *(const uint4*)(W + (size_t)(n0 + er)  * DD + kb + ec);
    wr1 = *(const uint4*)(W + (size_t)(n0 + er2) * DD + kb + ec2);
  };
  auto store_lds = [&]() {
    *(uint4*)(&Xs[er  * LDP + ec])  = xr0;
    *(uint4*)(&Xs[er2 * LDP + ec2]) = xr1;
    *(uint4*)(&Ws[er  * LDP + ec])  = wr0;
    *(uint4*)(&Ws[er2 * LDP + ec2]) = wr1;
  };

  v8f acc[2][4] = {};

  load_regs(0);
  store_lds();
  __syncthreads();

  for (int kb = 0; kb < DD; kb += 32) {
    bool more = (kb + 32) < DD;
    if (more) load_regs(kb + 32);  // overlap next-tile loads with WMMAs

    v16h af0 = load_frag(&Xs[(mq)      * LDP], LDP);
    v16h af1 = load_frag(&Xs[(mq + 16) * LDP], LDP);
#pragma unroll
    for (int t = 0; t < 4; ++t) {
      v16h bf = load_frag(&Ws[(nq + t * 16) * LDP], LDP);
      acc[0][t] = wmma_f16(af0, bf, acc[0][t]);
      acc[1][t] = wmma_f16(af1, bf, acc[1][t]);
    }
    __syncthreads();
    if (more) store_lds();
    __syncthreads();
  }

  int h = (n0 + nq) / DHD;  // 64-col wave tile is exactly one head

  if (z < 2) {  // RoPE on Q and K via table
#pragma unroll
    for (int a = 0; a < 2; ++a) {
#pragma unroll
      for (int t = 0; t < 2; ++t) {
        int d = t * 16 + l15;  // 0..31, partner d+32 lives in acc[a][t+2]
#pragma unroll
        for (int r = 0; r < 8; ++r) {
          int m = m0 + mq + a * 16 + rbase + r;
          int s = m & (SS - 1);
          float2 cssn = rtab[s * 32 + d];
          float x1 = acc[a][t][r], x2 = acc[a][t + 2][r];
          acc[a][t][r]     = x1 * cssn.x - x2 * cssn.y;
          acc[a][t + 2][r] = x2 * cssn.x + x1 * cssn.y;
        }
      }
    }
  }

  _Float16* Out = (z == 0) ? Qo : (z == 1) ? Ko : Vo;
#pragma unroll
  for (int a = 0; a < 2; ++a) {
#pragma unroll
    for (int t = 0; t < 4; ++t) {
      int d = t * 16 + l15;
#pragma unroll
      for (int r = 0; r < 8; ++r) {
        int m = m0 + mq + a * 16 + rbase + r;
        int s = m & (SS - 1), b = m >> 11;
        Out[((size_t)((b * HH + h) * SS + s)) * DHD + d] = (_Float16)acc[a][t][r];
      }
    }
  }
}

// ---------------------------------------------------------------------------
// Flash attention, software-pipelined key/value tiles.
// Mask simplifies to: row i<phase -> causal; i>=phase -> attends to all keys.
// grid = (S/128, B*H); block 256 (8 waves); wave owns 16 query rows.
// ---------------------------------------------------------------------------
__global__ __launch_bounds__(256) void flash_attn(
    const _Float16* __restrict__ Q, const _Float16* __restrict__ K,
    const _Float16* __restrict__ V, _Float16* __restrict__ A,
    const int* __restrict__ p_phase) {
  const int LDP = 72;
  __shared__ _Float16 Ks[64 * LDP];      // K tile [j][d]
  __shared__ _Float16 Vt[64 * LDP];      // V tile transposed [d][j]
  __shared__ _Float16 Ps[8 * 16 * LDP];  // per-wave P staging

  int m0 = blockIdx.x * 128;
  int bh = blockIdx.y;
  int b  = bh / HH, h = bh % HH;
  int phase = *p_phase;
  int tid = threadIdx.x, wave = tid >> 5, lane = tid & 31;
  int l15 = lane & 15, rbase = (lane & 16) ? 8 : 0;

  const _Float16* Qb = Q + (size_t)bh * SS * DHD;
  const _Float16* Kb = K + (size_t)bh * SS * DHD;
  const _Float16* Vb = V + (size_t)bh * SS * DHD;

  int mw = m0 + wave * 16;
  v16h aq0 = load_frag(Qb + (size_t)mw * DHD, DHD);       // d 0..31
  v16h aq1 = load_frag(Qb + (size_t)mw * DHD + 32, DHD);  // d 32..63

  v8f accO[4] = {};
  float mrow[8], lrow[8];
#pragma unroll
  for (int r = 0; r < 8; ++r) { mrow[r] = -3.0e38f; lrow[r] = 0.0f; }

  int imax = m0 + 127;
  int jend = (imax >= phase) ? SS : (imax + 1);  // multiple of 64

  // --- tile staging registers ---
  int kr  = (tid * 8) >> 6,         kc  = (tid * 8) & 63;
  int kr2 = ((tid + 256) * 8) >> 6, kc2 = ((tid + 256) * 8) & 63;
  int vj  = tid >> 2, vd = (tid & 3) * 16;
  uint4 kg0, kg1, vg0, vg1;
  auto load_tile_regs = [&](int jt) {
    kg0 = *(const uint4*)(Kb + (size_t)(jt + kr)  * DHD + kc);
    kg1 = *(const uint4*)(Kb + (size_t)(jt + kr2) * DHD + kc2);
    vg0 = *(const uint4*)(Vb + (size_t)(jt + vj) * DHD + vd);
    vg1 = *(const uint4*)(Vb + (size_t)(jt + vj) * DHD + vd + 8);
  };
  auto store_tile_lds = [&]() {
    *(uint4*)(&Ks[kr  * LDP + kc])  = kg0;
    *(uint4*)(&Ks[kr2 * LDP + kc2]) = kg1;
    _Float16 hv[16];
    *(uint4*)&hv[0] = vg0;
    *(uint4*)&hv[8] = vg1;
#pragma unroll
    for (int i = 0; i < 16; ++i) Vt[(vd + i) * LDP + vj] = hv[i];
  };

  load_tile_regs(0);
  store_tile_lds();
  __syncthreads();

  for (int jt = 0; jt < jend; jt += 64) {
    bool more = (jt + 64) < jend;
    if (more) load_tile_regs(jt + 64);  // overlap with this tile's compute
    {  // hint-prefetch the tile after next (global_prefetch_b8)
      int jpf = jt + 128;
      if (jpf + 64 <= SS) {
        __builtin_prefetch((const void*)(Kb + (size_t)jpf * DHD), 0, 1);
        __builtin_prefetch((const void*)(Vb + (size_t)jpf * DHD), 0, 1);
      }
    }

    // ---- scores: S = Q K^T ----
    v8f sc[4];
#pragma unroll
    for (int t = 0; t < 4; ++t) {
      v16h bk0 = load_frag(&Ks[(t * 16) * LDP], LDP);
      v16h bk1 = load_frag(&Ks[(t * 16) * LDP + 32], LDP);
      v8f c0 = {};
      c0 = wmma_f16(aq0, bk0, c0);
      c0 = wmma_f16(aq1, bk1, c0);
      sc[t] = c0;
    }

    // ---- mask + online softmax (f32 state) ----
#pragma unroll
    for (int r = 0; r < 8; ++r) {
      int i = mw + rbase + r;
      float mx = -3.0e38f;
#pragma unroll
      for (int t = 0; t < 4; ++t) {
        int j = jt + t * 16 + l15;
        float v = sc[t][r] * SCALE;
        bool ok = (j <= i) || (i >= phase);
        v = ok ? v : -3.0e38f;
        sc[t][r] = v;
        mx = fmaxf(mx, v);
      }
#pragma unroll
      for (int msk = 1; msk < 16; msk <<= 1) mx = fmaxf(mx, __shfl_xor(mx, msk, 32));
      float mnew = fmaxf(mrow[r], mx);
      float resc = __expf(mrow[r] - mnew);
      lrow[r] *= resc;
#pragma unroll
      for (int t = 0; t < 4; ++t) accO[t][r] *= resc;
      float rs = 0.0f;
#pragma unroll
      for (int t = 0; t < 4; ++t) {
        float v  = sc[t][r];
        float pv = (v < -1.0e37f) ? 0.0f : __expf(v - mnew);
        sc[t][r] = pv;
        rs += pv;
      }
#pragma unroll
      for (int msk = 1; msk < 16; msk <<= 1) rs += __shfl_xor(rs, msk, 32);
      lrow[r] += rs;
      mrow[r] = mnew;
    }

    // ---- P -> LDS (per-wave), C-layout -> A-layout reshape ----
    _Float16* Pw = &Ps[wave * 16 * LDP];
#pragma unroll
    for (int t = 0; t < 4; ++t)
#pragma unroll
      for (int r = 0; r < 8; ++r)
        Pw[(rbase + r) * LDP + t * 16 + l15] = (_Float16)sc[t][r];
    __syncthreads();

    // ---- O += P V ----
    v16h ap0 = load_frag(Pw, LDP);
    v16h ap1 = load_frag(Pw + 32, LDP);
#pragma unroll
    for (int t = 0; t < 4; ++t) {
      v16h bv0 = load_frag(&Vt[(t * 16) * LDP], LDP);
      v16h bv1 = load_frag(&Vt[(t * 16) * LDP + 32], LDP);
      accO[t] = wmma_f16(ap0, bv0, accO[t]);
      accO[t] = wmma_f16(ap1, bv1, accO[t]);
    }
    __syncthreads();           // everyone done reading Ks/Vt
    if (more) store_tile_lds();
    __syncthreads();           // next tile resident
  }

  // ---- normalize + store token-major f16 [B][S][H*64] ----
#pragma unroll
  for (int t = 0; t < 4; ++t) {
    int d = t * 16 + l15;
#pragma unroll
    for (int r = 0; r < 8; ++r) {
      int i = mw + rbase + r;
      float o = accO[t][r] / fmaxf(lrow[r], 1e-20f);
      A[((size_t)(b * SS + i)) * NQK + h * DHD + d] = (_Float16)o;
    }
  }
}

// ---------------------------------------------------------------------------
// Output projection: Y(f32) = A(f16) @ Wo ; same pipelined 128x128 tiling.
// ---------------------------------------------------------------------------
__global__ __launch_bounds__(256) void gemm_out(
    const _Float16* __restrict__ Ain, const _Float16* __restrict__ Wt,
    float* __restrict__ Y) {
  const int LDP = 40;
  __shared__ _Float16 Xs[128 * LDP];
  __shared__ _Float16 Ws[128 * LDP];

  int m0 = blockIdx.x * 128;
  int n0 = blockIdx.y * 128;
  int tid = threadIdx.x;
  int wave = tid >> 5, lane = tid & 31;
  int l15 = lane & 15;
  int rbase = (lane & 16) ? 8 : 0;
  int mq = (wave & 3) * 32;
  int nq = (wave >> 2) * 64;

  int er = (tid * 8) >> 5, ec = (tid * 8) & 31;
  int er2 = ((tid + 256) * 8) >> 5, ec2 = ((tid + 256) * 8) & 31;

  uint4 xr0, xr1, wr0, wr1;
  auto load_regs = [&](int kb) {
    xr0 = *(const uint4*)(Ain + (size_t)(m0 + er)  * DD + kb + ec);
    xr1 = *(const uint4*)(Ain + (size_t)(m0 + er2) * DD + kb + ec2);
    wr0 = *(const uint4*)(Wt + (size_t)(n0 + er)  * DD + kb + ec);
    wr1 = *(const uint4*)(Wt + (size_t)(n0 + er2) * DD + kb + ec2);
  };
  auto store_lds = [&]() {
    *(uint4*)(&Xs[er  * LDP + ec])  = xr0;
    *(uint4*)(&Xs[er2 * LDP + ec2]) = xr1;
    *(uint4*)(&Ws[er  * LDP + ec])  = wr0;
    *(uint4*)(&Ws[er2 * LDP + ec2]) = wr1;
  };

  v8f acc[2][4] = {};

  load_regs(0);
  store_lds();
  __syncthreads();

  for (int kb = 0; kb < DD; kb += 32) {
    bool more = (kb + 32) < DD;
    if (more) load_regs(kb + 32);

    v16h af0 = load_frag(&Xs[(mq)      * LDP], LDP);
    v16h af1 = load_frag(&Xs[(mq + 16) * LDP], LDP);
#pragma unroll
    for (int t = 0; t < 4; ++t) {
      v16h bf = load_frag(&Ws[(nq + t * 16) * LDP], LDP);
      acc[0][t] = wmma_f16(af0, bf, acc[0][t]);
      acc[1][t] = wmma_f16(af1, bf, acc[1][t]);
    }
    __syncthreads();
    if (more) store_lds();
    __syncthreads();
  }

#pragma unroll
  for (int a = 0; a < 2; ++a) {
#pragma unroll
    for (int t = 0; t < 4; ++t) {
      int n = n0 + nq + t * 16 + l15;
#pragma unroll
      for (int r = 0; r < 8; ++r) {
        int m = m0 + mq + a * 16 + rbase + r;
        Y[(size_t)m * DD + n] = acc[a][t][r];
      }
    }
  }
}

// ---------------------------------------------------------------------------
// Launch: cvt + rope table -> QKV+RoPE -> flash attention -> out projection
// Workspace (f16 unless noted):
//   X16 6.29MB | Wt 4.72MB | ropeTab(f32x2) 0.5MB | Q/K/V 6.29MB ea | A16 6.29MB
//   total ~36.7 MB
// ---------------------------------------------------------------------------
extern "C" void kernel_launch(void* const* d_in, const int* in_sizes, int n_in,
                              void* d_out, int out_size, void* d_ws,
                              size_t ws_size, hipStream_t stream) {
  const float* x  = (const float*)d_in[0];
  const float* wq = (const float*)d_in[1];
  const float* wk = (const float*)d_in[2];
  const float* wv = (const float*)d_in[3];
  const float* wo = (const float*)d_in[4];
  const int* phase = (const int*)d_in[5];
  float* out = (float*)d_out;

  char* ws = (char*)d_ws;
  size_t off = 0;
  _Float16* X16 = (_Float16*)(ws + off); off += (size_t)MTOT * DD * 2;
  _Float16* Wt  = (_Float16*)(ws + off); off += (size_t)4 * DD * DD * 2;
  float2*   rT  = (float2*)(ws + off);   off += (size_t)SS * 32 * sizeof(float2);
  const size_t hd = (size_t)BB * HH * SS * DHD;
  _Float16* Q16 = (_Float16*)(ws + off); off += hd * 2;
  _Float16* K16 = (_Float16*)(ws + off); off += hd * 2;
  _Float16* V16 = (_Float16*)(ws + off); off += hd * 2;
  _Float16* A16 = (_Float16*)(ws + off); off += (size_t)MTOT * DD * 2;

  {
    int n = MTOT * DD;
    cvt_x<<<dim3((n / 4 + 255) / 256), dim3(256), 0, stream>>>(x, X16, n);
  }
  {
    int n = 4 * DD * DD;
    cvt_w<<<dim3((n + 255) / 256), dim3(256), 0, stream>>>(wq, wk, wv, wo, Wt);
  }
  rope_tab_k<<<dim3((SS * 32 + 255) / 256), dim3(256), 0, stream>>>(rT);

  gemm_qkv_rope<<<dim3(MTOT / 128, NQK / 128, 3), dim3(256), 0, stream>>>(
      X16, Wt, rT, Q16, K16, V16);
  flash_attn<<<dim3(SS / 128, BB * HH), dim3(256), 0, stream>>>(
      Q16, K16, V16, A16, phase);
  gemm_out<<<dim3(MTOT / 128, DD / 128), dim3(256), 0, stream>>>(
      A16, Wt + (size_t)3 * DD * DD, out);
}
